// CTAG_87273735454729
// MI455X (gfx1250) — matrix operand order repaired and checked by reading, not compiled, not run on hardware
//
#include <hip/hip_runtime.h>
#include <hip/hip_bf16.h>
#include <math.h>

// ---------------- problem constants ----------------
#define B_    2
#define T_    20
#define V_    256
#define PTS   (B_*T_*V_)      // 10240 points total
#define PPB   (T_*V_)         // 5120 points per batch
#define HID   4
#define DQ    8
#define OUTD  5
#define CC    256
#define CMAP  2048
#define HMAP  32
#define WMAP  32
#define NPOS  (HMAP*WMAP)     // 1024
#define THR   0.5f
#define IMG_  512.0f
#define QTILES (PPB/16)       // 320 q-tiles per batch
#define ROWT   (B_*QTILES)    // 640 row-tiles total

// ---------------- WMMA types ----------------
typedef __attribute__((ext_vector_type(16))) __bf16 v16bf;
typedef __attribute__((ext_vector_type(8)))  float  v8f;
union BF16Frag { v16bf v; unsigned u[8]; unsigned short s[16]; };

__device__ __forceinline__ unsigned short f2bf(float f) {
  unsigned u = __float_as_uint(f);
  return (unsigned short)((u + 0x7FFFu + ((u >> 16) & 1u)) >> 16);
}
__device__ __forceinline__ float sigm(float x) { return 1.0f / (1.0f + __expf(-x)); }

// ---------------- workspace layout (bytes) ----------------
#define OFF_WFCA 0u                                   // 16*2048 bf16       = 65536
#define OFF_MT   (OFF_WFCA + 16u*2048u*2u)            // 1024*2048 bf16     = 4 MiB
#define OFF_GB   (OFF_MT   + 1024u*2048u*2u)          // 4 f32 (pad 256)
#define OFF_G4   (OFF_GB   + 256u)                    // 4*1024 f32
#define OFF_QP   (OFF_G4   + 4u*1024u*4u)             // 10240*32 bf16
#define OFF_KP   (OFF_QP   + (unsigned)PTS*32u*2u)    // 10240*32 bf16
#define OFF_VT   (OFF_KP   + (unsigned)PTS*32u*2u)    // 16*10240 bf16 (transposed V)
#define OFF_ATT  (OFF_VT   + 16u*(unsigned)PTS*2u)    // 10240*8 f32

// ============================================================
// Prep 1: fused weight  Wfc[h][k] = sum_c W_fuse[h][4+c]*W_comp[c][k]
// stored bf16, padded to 16 rows (rows 4..15 zero) for the WMMA A tile.
// ============================================================
__global__ void k_prep_wfc(const float* __restrict__ W_fuse,
                           const float* __restrict__ W_comp,
                           unsigned short* __restrict__ wfca) {
  int idx = blockIdx.x * 256 + threadIdx.x;          // 0..32767
  int h = idx >> 11, k = idx & 2047;
  float acc = 0.0f;
  if (h < HID) {
    for (int c = 0; c < CC; ++c)
      acc += W_fuse[h * (HID + CC) + HID + c] * W_comp[c * CMAP + k];
  }
  wfca[h * CMAP + k] = f2bf(acc);
}

// Prep 2: gbias[h] = sum_c W_fuse[h][4+c]*b_comp[c]  (per-position constant)
__global__ void k_prep_gbias(const float* __restrict__ W_fuse,
                             const float* __restrict__ b_comp,
                             float* __restrict__ gb) {
  int h = threadIdx.x;
  if (h < HID) {
    float a = 0.0f;
    for (int c = 0; c < CC; ++c) a += W_fuse[h * (HID + CC) + HID + c] * b_comp[c];
    gb[h] = a;
  }
}

// Prep 3: metadata (C,H,W) f32 -> transposed bf16 MT[hw][c] (B-frag loads contiguous in K)
__global__ void k_prep_mt(const float* __restrict__ metadata,
                          unsigned short* __restrict__ mt) {
  int idx = blockIdx.x * 256 + threadIdx.x;          // 0..2097151
  int hw = idx >> 11, c = idx & 2047;
  mt[idx] = f2bf(metadata[c * NPOS + hw]);
}

// ============================================================
// GEMM: G4(16x1024, rows 0..3 valid) = WfcA(16x2048) @ M(2048x1024) + gbias
// One wave per 16x16 output tile; K loop in steps of 32 via v_wmma_f32_16x16x32_bf16.
// ============================================================
__global__ void k_gemm_g(const unsigned short* __restrict__ wfca,
                         const unsigned short* __restrict__ mt,
                         const float* __restrict__ gb,
                         float* __restrict__ g4) {
  int lane = threadIdx.x & 31;
  int wid  = blockIdx.x * (blockDim.x >> 5) + (threadIdx.x >> 5);   // N tile 0..63
  int half = lane >> 4, r16 = lane & 15;
  v8f acc = {0.f,0.f,0.f,0.f,0.f,0.f,0.f,0.f};
  for (int kk = 0; kk < CMAP; kk += 32) {
    BF16Frag a, b;
#pragma unroll
    for (int j = 0; j < 8; ++j) {
      int kba = 2*j + 16*half;                                      // A 16x32 layout
      a.u[j] = *(const unsigned*)(wfca + r16 * CMAP + kk + kba);
      int kbb = 2*j + 16*half;                                      // B 32x16 layout
      b.u[j] = *(const unsigned*)(mt + (size_t)(wid*16 + r16) * CMAP + kk + kbb);
    }
    acc = __builtin_amdgcn_wmma_f32_16x16x32_bf16(false, a.v, false, b.v,
                                                  (short)0, acc, false, false);
  }
#pragma unroll
  for (int r = 0; r < 8; ++r) {
    int m = r + 8*half;                                             // C/D layout
    if (m < HID) g4[m * NPOS + wid*16 + r16] = acc[r] + gb[m];
  }
}

// ============================================================
// Per-point: pos-encode + LSTM step + bilinear sample of fused 4-ch map
// + fuse + Q/K/V projections.
// Q,K stored bf16 row-major zero-padded to K=32; V stored TRANSPOSED
// Vt[d][point] (d padded to 16, rows 8..15 zero) so the attention B-frag
// K-dim pairs (two consecutive q) are contiguous dword loads.
// ============================================================
__global__ void k_points(const float* __restrict__ x,
                         const float* __restrict__ abs_coords,
                         const float* __restrict__ W_ih,
                         const float* __restrict__ b_ih,
                         const float* __restrict__ b_hh,
                         const float* __restrict__ W_fuse,
                         const float* __restrict__ b_fuse,
                         const float* __restrict__ W_fc,  const float* __restrict__ b_fc,
                         const float* __restrict__ W_fc2, const float* __restrict__ b_fc2,
                         const float* __restrict__ W_fc3, const float* __restrict__ b_fc3,
                         const float* __restrict__ g4,
                         unsigned short* __restrict__ qp,
                         unsigned short* __restrict__ kp,
                         unsigned short* __restrict__ vt) {
  int i = blockIdx.x * 256 + threadIdx.x;              // < 10240
  int b = i / PPB, rem = i % PPB, t = rem / V_, n = rem % V_;

  // pos encode (c=2 -> div = 1.0)
  float xr0 = x[i*2+0] + sinf((float)t);
  float xr1 = x[i*2+1] + cosf((float)t);

  // single-step LSTM, h0=c0=0: gates i,f,g,o ; f unused (c0=0)
  float Xl[HID];
#pragma unroll
  for (int h = 0; h < HID; ++h) {
    float gi = W_ih[h*2]*xr0      + W_ih[h*2+1]*xr1      + b_ih[h]    + b_hh[h];
    float gg = W_ih[(8+h)*2]*xr0  + W_ih[(8+h)*2+1]*xr1  + b_ih[8+h]  + b_hh[8+h];
    float go = W_ih[(12+h)*2]*xr0 + W_ih[(12+h)*2+1]*xr1 + b_ih[12+h] + b_hh[12+h];
    float cs = sigm(gi) * tanhf(gg);
    Xl[h] = sigm(go) * tanhf(cs);
  }

  // bilinear grid-sample of fused 4-channel map (zero padding, align_corners=False)
  float xc = abs_coords[((b*2+0)*T_ + t)*V_ + n];
  float yc = abs_coords[((b*2+1)*T_ + t)*V_ + n];
  float gx = 2.0f*xc/IMG_ - 1.0f, gy = 2.0f*yc/IMG_ - 1.0f;
  float fx = ((gx + 1.0f)*WMAP - 1.0f)*0.5f;
  float fy = ((gy + 1.0f)*HMAP - 1.0f)*0.5f;
  float x0f = floorf(fx), y0f = floorf(fy);
  float wx1 = fx - x0f, wx0 = 1.0f - wx1, wy1 = fy - y0f, wy0 = 1.0f - wy1;
  int x0i = (int)x0f, y0i = (int)y0f;
  float loc[HID] = {0.f,0.f,0.f,0.f};
#pragma unroll
  for (int dy = 0; dy < 2; ++dy)
#pragma unroll
    for (int dx = 0; dx < 2; ++dx) {
      int yi = y0i + dy, xi = x0i + dx;
      float w = (dy ? wy1 : wy0) * (dx ? wx1 : wx0);
      if (yi >= 0 && yi < HMAP && xi >= 0 && xi < WMAP) {
        int pos = yi*WMAP + xi;
#pragma unroll
        for (int h = 0; h < HID; ++h) loc[h] += w * g4[h*NPOS + pos];
      }
    }

  // fuse (visual part already folded into g4) + projections
  float Q[DQ], K8[DQ], V8[DQ], Xf[HID];
#pragma unroll
  for (int h = 0; h < HID; ++h) {
    float a = b_fuse[h] + loc[h];
#pragma unroll
    for (int j = 0; j < HID; ++j) a += Xl[j] * W_fuse[h*(HID+CC) + j];
    Xf[h] = a;
  }
#pragma unroll
  for (int d = 0; d < DQ; ++d) {
    float a = b_fc[d];
#pragma unroll
    for (int h = 0; h < HID; ++h) a += Xf[h] * W_fc[d*HID + h];
    Q[d]  = a;
    K8[d] = b_fc2[d] + xr0*W_fc2[d*2] + xr1*W_fc2[d*2+1];
    V8[d] = b_fc3[d] + xr0*W_fc3[d*2] + xr1*W_fc3[d*2+1];
  }
#pragma unroll
  for (int d = 0; d < 32; ++d) {
    qp[(size_t)i*32 + d] = (d < DQ) ? f2bf(Q[d])  : (unsigned short)0;
    kp[(size_t)i*32 + d] = (d < DQ) ? f2bf(K8[d]) : (unsigned short)0;
  }
#pragma unroll
  for (int d = 0; d < 16; ++d)                        // transposed, padded rows zero
    vt[(size_t)d*PTS + i] = (d < DQ) ? f2bf(V8[d]) : (unsigned short)0;
}

// ============================================================
// Fused attention: per wave one 16-row tile; loop q-tiles:
//   S = WMMA(Q, K^T)  ->  sigmoid  ->  LDS relayout  ->  O += WMMA(S, V)
// then threshold-relu, store f32.
// ============================================================
__global__ void k_attn(const unsigned short* __restrict__ qp,
                       const unsigned short* __restrict__ kp,
                       const unsigned short* __restrict__ vt,
                       float* __restrict__ att) {
  __shared__ unsigned short sS[8][16][16];             // per-wave S staging
  int lane = threadIdx.x & 31, w = threadIdx.x >> 5;
  int wg = blockIdx.x * 8 + w;                          // 0..639
  int batch = wg / QTILES, pt = wg % QTILES;
  int half = lane >> 4, r16 = lane & 15;

  BF16Frag qa;                                          // A frag: Q rows (K padded to 32)
  int prow = batch*PPB + pt*16 + r16;
#pragma unroll
  for (int j = 0; j < 8; ++j) {
    int kb = 2*j + 16*half;
    qa.u[j] = *(const unsigned*)(qp + (size_t)prow*32 + kb);
  }

  v8f O = {0.f,0.f,0.f,0.f,0.f,0.f,0.f,0.f};
  for (int qt = 0; qt < QTILES; ++qt) {
    int qbase = batch*PPB + qt*16;
    BF16Frag kf;                                        // B frag: K^T (kd, q)
#pragma unroll
    for (int j = 0; j < 8; ++j) {
      int k2 = 2*j + 16*half;
      kf.u[j] = *(const unsigned*)(kp + (size_t)(qbase + r16)*32 + k2);
    }
    v8f z = {0.f,0.f,0.f,0.f,0.f,0.f,0.f,0.f};
    v8f S = __builtin_amdgcn_wmma_f32_16x16x32_bf16(false, qa.v, false, kf.v,
                                                    (short)0, z, false, false);
#pragma unroll
    for (int r = 0; r < 8; ++r)                         // sigmoid, stage to LDS
      sS[w][r + 8*half][r16] = f2bf(sigm(S[r]));
    asm volatile("s_wait_dscnt 0x0" ::: "memory");      // intra-wave LDS RAW

    BF16Frag sa, va;                                    // S as A (K 16..31 zero)
#pragma unroll
    for (int j = 0; j < 8; ++j) {
      if (!half) sa.u[j] = *(const unsigned*)&sS[w][r16][2*j];
      else       sa.u[j] = 0u;
    }
#pragma unroll
    for (int j = 0; j < 8; ++j) {                       // V as B: Vt[d][q], q-pairs contiguous
      unsigned vv = *(const unsigned*)(vt + (size_t)r16*PTS + qbase + 2*j);
      va.u[j] = half ? 0u : vv;                         // K rows 16..31 zero (select, EXEC full)
    }
    O = __builtin_amdgcn_wmma_f32_16x16x32_bf16(false, sa.v, false, va.v,
                                                (short)0, O, false, false);
  }
#pragma unroll
  for (int r = 0; r < 8; ++r) {
    int p = pt*16 + r + 8*half, d = r16;
    if (d < DQ) {
      float o = O[r];
      o = (o > THR) ? o : 0.0f;                         // threshold relu
      att[(size_t)(batch*PPB + p)*DQ + d] = o;
    }
  }
}

// ============================================================
// Final 5x8 projection + transpose to (b, OUT_DIM, t, n)
// ============================================================
__global__ void k_final(const float* __restrict__ att,
                        const float* __restrict__ W_out,
                        const float* __restrict__ b_out,
                        float* __restrict__ out) {
  int idx = blockIdx.x * 256 + threadIdx.x;            // < 51200
  int n = idx & 255;
  int t = (idx >> 8) % T_;
  int od = (idx / (256*T_)) % OUTD;
  int b = idx / (256*T_*OUTD);
  int i = (b*T_ + t)*V_ + n;
  float a = b_out[od];
#pragma unroll
  for (int d = 0; d < DQ; ++d) a += att[(size_t)i*DQ + d] * W_out[od*DQ + d];
  out[idx] = a;
}

// ============================================================
extern "C" void kernel_launch(void* const* d_in, const int* in_sizes, int n_in,
                              void* d_out, int out_size, void* d_ws, size_t ws_size,
                              hipStream_t stream) {
  (void)in_sizes; (void)n_in; (void)out_size; (void)ws_size;
  const float* x          = (const float*)d_in[0];
  const float* abs_coords = (const float*)d_in[1];
  const float* metadata   = (const float*)d_in[2];
  const float* W_ih       = (const float*)d_in[3];
  // d_in[4] = W_hh : unused (h0 = 0)
  const float* b_ih       = (const float*)d_in[5];
  const float* b_hh       = (const float*)d_in[6];
  const float* W_comp     = (const float*)d_in[7];
  const float* b_comp     = (const float*)d_in[8];
  const float* W_fuse     = (const float*)d_in[9];
  const float* b_fuse     = (const float*)d_in[10];
  const float* W_fc       = (const float*)d_in[11];
  const float* b_fc       = (const float*)d_in[12];
  const float* W_fc2      = (const float*)d_in[13];
  const float* b_fc2      = (const float*)d_in[14];
  const float* W_fc3      = (const float*)d_in[15];
  const float* b_fc3      = (const float*)d_in[16];
  const float* W_out      = (const float*)d_in[17];
  const float* b_out      = (const float*)d_in[18];
  float* out = (float*)d_out;

  char* ws = (char*)d_ws;
  unsigned short* wfca = (unsigned short*)(ws + OFF_WFCA);
  unsigned short* mt   = (unsigned short*)(ws + OFF_MT);
  float*          gb   = (float*)(ws + OFF_GB);
  float*          g4   = (float*)(ws + OFF_G4);
  unsigned short* qp   = (unsigned short*)(ws + OFF_QP);
  unsigned short* kp   = (unsigned short*)(ws + OFF_KP);
  unsigned short* vt   = (unsigned short*)(ws + OFF_VT);
  float*          att  = (float*)(ws + OFF_ATT);

  k_prep_wfc  <<<128,  256, 0, stream>>>(W_fuse, W_comp, wfca);
  k_prep_gbias<<<1,    32,  0, stream>>>(W_fuse, b_comp, gb);
  k_prep_mt   <<<8192, 256, 0, stream>>>(metadata, mt);
  k_gemm_g    <<<8,    256, 0, stream>>>(wfca, mt, gb, g4);
  k_points    <<<40,   256, 0, stream>>>(x, abs_coords, W_ih, b_ih, b_hh,
                                         W_fuse, b_fuse, W_fc, b_fc,
                                         W_fc2, b_fc2, W_fc3, b_fc3,
                                         g4, qp, kp, vt);
  k_attn      <<<80,   256, 0, stream>>>(qp, kp, vt, att);
  k_final     <<<200,  256, 0, stream>>>(att, W_out, b_out, out);
}
